// BidirRNNLayer_59665685676324
// MI455X (gfx1250) — compile-verified, matrix operand order checked by reading, not run on hardware
//
#include <hip/hip_runtime.h>
#include <hip/hip_bf16.h>

// Problem constants (reference: T, B, D, H = 256, 32, 768, 768)
#define Tn  256
#define Bn  32
#define Dn  768
#define Hn  768
#define G3H (3 * Hn)      // 2304 gate columns (r | z | n)
#define NBLK (Hn / 64)    // 12 persistent blocks per direction

typedef __bf16 bf16_t;
typedef __attribute__((ext_vector_type(16))) __bf16 v16bf;
typedef __attribute__((ext_vector_type(8)))  __bf16 bf8v;
typedef __attribute__((ext_vector_type(8)))  float  v8f;

// gfx1250 async global->LDS path (ASYNCcnt), with compile-safe fallback.
#if defined(__HIP_DEVICE_COMPILE__) && \
    __has_builtin(__builtin_amdgcn_global_load_async_to_lds_b128) && \
    __has_builtin(__builtin_amdgcn_s_wait_asynccnt)
#define USE_ASYNC_LDS 1
#define AS1 __attribute__((address_space(1)))
#define AS3 __attribute__((address_space(3)))
typedef int v4i_vs __attribute__((vector_size(16)));   // builtin's b128 unit
#else
#define USE_ASYNC_LDS 0
#endif

// ---------------------------------------------------------------------------
// WMMA bf16 fragment loader (ISA §7.12.2, 16-bit A-matrix 16x32):
//   lane l: row = row0 + (l & 15), khalf = l >> 4
//   elems [0..7]  hold K = k0 +      8*khalf + [0..8)
//   elems [8..15] hold K = k0 + 16 + 8*khalf + [0..8)
// B (32x16, B[k][n] = W[n][k], W row-major) has the identical pattern over
// W's rows, so one loader serves both operands. Two b128 loads per lane.
// ---------------------------------------------------------------------------
__device__ __forceinline__ v16bf load_tile_frag(const bf16_t* __restrict__ src,
                                                int ld, int row0, int k0, int lane) {
  const int r  = row0 + (lane & 15);
  const int kh = (lane >> 4) & 1;
  const bf16_t* p = src + (size_t)r * ld + (k0 + kh * 8);
  bf8v lo = *(const bf8v*)(p);
  bf8v hi = *(const bf8v*)(p + 16);
  v16bf f;
#pragma unroll
  for (int i = 0; i < 8; ++i) { f[i] = lo[i]; f[i + 8] = hi[i]; }
  return f;
}

__device__ __forceinline__ v8f wmma_bf16(v16bf a, v16bf b, v8f c) {
  return __builtin_amdgcn_wmma_f32_16x16x32_bf16(false, a, false, b, (short)0, c,
                                                 false, false);
}

// ---------------------------------------------------------------------------
// Setup kernels
// ---------------------------------------------------------------------------
__global__ __launch_bounds__(256) void f32_to_bf16_kernel(const float* __restrict__ s,
                                                          bf16_t* __restrict__ d, int n) {
  int i = blockIdx.x * 256 + threadIdx.x;
  if (i < n) d[i] = (bf16_t)s[i];
}

// h0 -> double-buffered h (fp32 + bf16), buffer 0; zero the grid barriers.
__global__ __launch_bounds__(256) void init_h_kernel(const float* __restrict__ h0,
                                                     float* __restrict__ hF,
                                                     bf16_t* __restrict__ hB,
                                                     unsigned* __restrict__ bar) {
  int i = blockIdx.x * 256 + threadIdx.x;
  if (i < 4) bar[i] = 0u;
  if (i < 2 * Bn * Hn) {
    int dir = i / (Bn * Hn), r = i % (Bn * Hn);
    float v = h0[i];
    hF[(size_t)dir * 2 * Bn * Hn + r] = v;        // [dir][buf=0][...]
    hB[(size_t)dir * 2 * Bn * Hn + r] = (bf16_t)v;
  }
}

// final hidden state (after 256 steps the live buffer is 0) -> d_out tail
__global__ __launch_bounds__(256) void final_h_kernel(const float* __restrict__ hF,
                                                      float* __restrict__ dst) {
  int i = blockIdx.x * 256 + threadIdx.x;
  if (i < 2 * Bn * Hn) {
    int dir = i / (Bn * Hn), r = i % (Bn * Hn);
    dst[i] = hF[(size_t)dir * 2 * Bn * Hn + r];
  }
}

// ---------------------------------------------------------------------------
// Bulk input projection: gx[dir] = x_bf16 @ Wih[dir]^T + bih[dir]
// M = 8192, N = 2304, K = 768. Block tile 64x256, wave tile 32x64
// (2 A-frags + 4 B-frags -> 8 WMMAs per K-slab: 2.7x better reuse than 16x16).
// ---------------------------------------------------------------------------
__global__ __launch_bounds__(256)
void gx_gemm_kernel(const bf16_t* __restrict__ xb,    // [T*B][D]
                    const bf16_t* __restrict__ wihB,  // [2][3H][D]
                    const float*  __restrict__ bih_f,
                    const float*  __restrict__ bih_b,
                    float* __restrict__ gx)           // [2][T*B][3H]
{
  const int dir = blockIdx.z;
  const bf16_t* w   = wihB + (size_t)dir * G3H * Dn;
  const float*  bih = dir ? bih_b : bih_f;
  const int lane = threadIdx.x & 31;
  const int wv   = threadIdx.x >> 5;
  const int m0 = blockIdx.x * 64  + (wv >> 2) * 32;
  const int n0 = blockIdx.y * 256 + (wv & 3) * 64;

  v8f acc[2][4] = {};
  for (int k0 = 0; k0 < Dn; k0 += 32) {
    __builtin_prefetch(xb + (size_t)(m0 + (lane & 15)) * Dn + k0 + 128, 0, 1);
    __builtin_prefetch(w  + (size_t)(n0 + (lane & 15)) * Dn + k0 + 128, 0, 1);
    v16bf a[2], b[4];
#pragma unroll
    for (int i = 0; i < 2; ++i) a[i] = load_tile_frag(xb, Dn, m0 + 16 * i, k0, lane);
#pragma unroll
    for (int j = 0; j < 4; ++j) b[j] = load_tile_frag(w, Dn, n0 + 16 * j, k0, lane);
#pragma unroll
    for (int i = 0; i < 2; ++i)
#pragma unroll
      for (int j = 0; j < 4; ++j)
        acc[i][j] = wmma_bf16(a[i], b[j], acc[i][j]);
  }

  const int kh = (lane >> 4) & 1;
  float* dst = gx + (size_t)dir * (Tn * Bn) * G3H;
#pragma unroll
  for (int i = 0; i < 2; ++i)
#pragma unroll
    for (int j = 0; j < 4; ++j) {
      const int n = n0 + 16 * j + (lane & 15);
      const float bias = bih[n];
#pragma unroll
      for (int v = 0; v < 8; ++v) {
        const int m = m0 + 16 * i + v + 8 * kh;   // C/D layout: VGPR v -> row v+8*kh
        dst[(size_t)m * G3H + n] = acc[i][j][v] + bias;
      }
    }
}

// ---------------------------------------------------------------------------
// Persistent GRU scan: one launch runs all 256 steps for both directions.
// Per direction: 12 blocks (64 hidden columns each) synchronized each step by
// a device-scope sense-reversal barrier (release/acquire atomics + s_sleep).
// h (32x768 bf16 = 48 KB) is staged into LDS each step with
// GLOBAL_LOAD_ASYNC_TO_LDS_B128 (ASYNCcnt) when the toolchain exposes it.
// ---------------------------------------------------------------------------
__global__ __launch_bounds__(256)
void gru_scan_kernel(const bf16_t* __restrict__ whhB,  // [2][3H][H]
                     const float* __restrict__ bhh_f,
                     const float* __restrict__ bhh_b,
                     const float* __restrict__ gx,     // [2][T*B][3H]
                     const int*   __restrict__ lengths,
                     float*  __restrict__ hF,          // [2][2][B][H]
                     bf16_t* __restrict__ hB,          // [2][2][B][H]
                     float*  __restrict__ out,         // [T][B][2H]
                     unsigned* __restrict__ bar)       // [dir*2 + {count,gen}]
{
  const int dir = blockIdx.z;
  const bf16_t* whh = whhB + (size_t)dir * G3H * Hn;
  const float*  bhh = dir ? bhh_b : bhh_f;
  unsigned* cnt = bar + dir * 2;
  unsigned* gen = bar + dir * 2 + 1;

  const int lane = threadIdx.x & 31;
  const int wv   = threadIdx.x >> 5;
  const int m0 = (wv >> 2) * 16;                  // batch tile (0 or 16)
  const int cn = blockIdx.x * 64 + (wv & 3) * 16; // hidden-column tile
  const int kh  = (lane >> 4) & 1;
  const int col = cn + (lane & 15);
  const float br = bhh[col], bz = bhh[Hn + col], bn = bhh[2 * Hn + col];

  int len[8];
#pragma unroll
  for (int v = 0; v < 8; ++v) len[v] = lengths[m0 + v + 8 * kh];

  __shared__ __align__(16) bf16_t sh[Bn * Hn];    // 48 KB (of 320 KB/WGP)

  for (int step = 0; step < Tn; ++step) {
    const int t    = dir ? (Tn - 1 - step) : step;
    const int bufI = step & 1, bufO = bufI ^ 1;
    const float*  hin  = hF + ((size_t)dir * 2 + bufI) * Bn * Hn;
    float*        hout = hF + ((size_t)dir * 2 + bufO) * Bn * Hn;
    const bf16_t* hinb = hB + ((size_t)dir * 2 + bufI) * Bn * Hn;
    bf16_t*       houtb= hB + ((size_t)dir * 2 + bufO) * Bn * Hn;
    const float*  gxd  = gx + ((size_t)dir * Tn * Bn + (size_t)t * Bn) * G3H;

#if USE_ASYNC_LDS
    {
      AS1 v4i_vs* g = (AS1 v4i_vs*)const_cast<bf16_t*>(hinb);
      AS3 v4i_vs* l = (AS3 v4i_vs*)sh;
      const int idx0 = threadIdx.x;               // 256 thr x 16B = 4 KB / sweep
#pragma unroll
      for (int i = 0; i < (Bn * Hn * 2) / 4096; ++i)   // 12 sweeps = 48 KB
        __builtin_amdgcn_global_load_async_to_lds_b128(g + idx0 + i * 256,
                                                       l + idx0 + i * 256, 0, 0);
      __builtin_amdgcn_s_wait_asynccnt(0);
    }
#else
    for (int i = threadIdx.x; i < (Bn * Hn) / 8; i += 256)
      ((bf8v*)sh)[i] = ((const bf8v*)hinb)[i];
#endif
    __syncthreads();

    v8f accR = {}, accZ = {}, accN = {};
    for (int k0 = 0; k0 < Hn; k0 += 32) {
      v16bf a  = load_tile_frag(sh, Hn, m0, k0, lane);        // ds_load_b128 x2
      v16bf bR = load_tile_frag(whh, Hn, 0 * Hn + cn, k0, lane);
      v16bf bZ = load_tile_frag(whh, Hn, 1 * Hn + cn, k0, lane);
      v16bf bN = load_tile_frag(whh, Hn, 2 * Hn + cn, k0, lane);
      accR = wmma_bf16(a, bR, accR);
      accZ = wmma_bf16(a, bZ, accZ);
      accN = wmma_bf16(a, bN, accN);
    }

#pragma unroll
    for (int v = 0; v < 8; ++v) {
      const int m = m0 + v + 8 * kh;              // batch row
      const bool live = t < len[v];
      // gx is streamed exactly once: non-temporal so it can't evict Whh from L2
      const float xr = __builtin_nontemporal_load(&gxd[(size_t)m * G3H + col]);
      const float xz = __builtin_nontemporal_load(&gxd[(size_t)m * G3H + Hn + col]);
      const float xn = __builtin_nontemporal_load(&gxd[(size_t)m * G3H + 2 * Hn + col]);
      const float r  = 1.f / (1.f + __expf(-(xr + accR[v] + br)));
      const float z  = 1.f / (1.f + __expf(-(xz + accZ[v] + bz)));
      const float pre = xn + r * (accN[v] + bn);
      const float e2  = __expf(2.f * pre);
      const float nn  = (e2 - 1.f) / (e2 + 1.f);  // tanh(pre)
      const float hold = hin[(size_t)m * Hn + col];
      float hnew = (1.f - z) * nn + z * hold;
      hnew = live ? hnew : hold;
      hout[(size_t)m * Hn + col]  = hnew;
      houtb[(size_t)m * Hn + col] = (bf16_t)hnew;
      __builtin_nontemporal_store(live ? hnew : 0.f,
          &out[((size_t)t * Bn + m) * (2 * Hn) + (size_t)dir * Hn + col]);
    }

    // ---- device-scope barrier across this direction's NBLK blocks ----
    __threadfence();          // make this thread's h stores device-visible
    __syncthreads();          // whole block arrived (and done reading sh)
    if (threadIdx.x == 0) {
      unsigned arrived = __hip_atomic_fetch_add(cnt, 1u, __ATOMIC_ACQ_REL,
                                                __HIP_MEMORY_SCOPE_AGENT);
      if (arrived == (unsigned)(NBLK - 1)) {
        __hip_atomic_store(cnt, 0u, __ATOMIC_RELAXED, __HIP_MEMORY_SCOPE_AGENT);
        __hip_atomic_fetch_add(gen, 1u, __ATOMIC_RELEASE, __HIP_MEMORY_SCOPE_AGENT);
      } else {
        while (__hip_atomic_load(gen, __ATOMIC_ACQUIRE,
                                 __HIP_MEMORY_SCOPE_AGENT) <= (unsigned)step)
          __builtin_amdgcn_s_sleep(1);
      }
    }
    __syncthreads();
  }
}

// ---------------------------------------------------------------------------
extern "C" void kernel_launch(void* const* d_in, const int* in_sizes, int n_in,
                              void* d_out, int out_size, void* d_ws, size_t ws_size,
                              hipStream_t stream) {
  const float* x     = (const float*)d_in[0];
  const float* h0    = (const float*)d_in[1];
  const float* wih_f = (const float*)d_in[2];
  const float* whh_f = (const float*)d_in[3];
  const float* bih_f = (const float*)d_in[4];
  const float* bhh_f = (const float*)d_in[5];
  const float* wih_b = (const float*)d_in[6];
  const float* whh_b = (const float*)d_in[7];
  const float* bih_b = (const float*)d_in[8];
  const float* bhh_b = (const float*)d_in[9];
  const int*   lengths = (const int*)d_in[10];
  float* out = (float*)d_out;

  const size_t XEL = (size_t)Tn * Bn * Dn;   // 6,291,456
  const size_t WEL = (size_t)G3H * Dn;       // 1,769,472

  // Workspace carve-up (all region sizes are multiples of 16 bytes).
  char* ws = (char*)d_ws;
  bf16_t*   xb   = (bf16_t*)ws;   ws += XEL * sizeof(bf16_t);
  bf16_t*   wihB = (bf16_t*)ws;   ws += 2 * WEL * sizeof(bf16_t);
  bf16_t*   whhB = (bf16_t*)ws;   ws += 2 * WEL * sizeof(bf16_t);
  float*    gx   = (float*)ws;    ws += (size_t)2 * Tn * Bn * G3H * sizeof(float);
  float*    hF   = (float*)ws;    ws += (size_t)2 * 2 * Bn * Hn * sizeof(float);
  bf16_t*   hB   = (bf16_t*)ws;   ws += (size_t)2 * 2 * Bn * Hn * sizeof(bf16_t);
  unsigned* bar  = (unsigned*)ws; ws += 4 * sizeof(unsigned);
  (void)ws_size; (void)n_in; (void)in_sizes; (void)out_size;

  // 1) bf16 conversions
  f32_to_bf16_kernel<<<(XEL + 255) / 256, 256, 0, stream>>>(x, xb, (int)XEL);
  f32_to_bf16_kernel<<<(WEL + 255) / 256, 256, 0, stream>>>(wih_f, wihB,       (int)WEL);
  f32_to_bf16_kernel<<<(WEL + 255) / 256, 256, 0, stream>>>(wih_b, wihB + WEL, (int)WEL);
  f32_to_bf16_kernel<<<(WEL + 255) / 256, 256, 0, stream>>>(whh_f, whhB,       (int)WEL);
  f32_to_bf16_kernel<<<(WEL + 255) / 256, 256, 0, stream>>>(whh_b, whhB + WEL, (int)WEL);

  // 2) h state + barrier init
  init_h_kernel<<<(2 * Bn * Hn + 255) / 256, 256, 0, stream>>>(h0, hF, hB, bar);

  // 3) hoisted input projection (fully parallel WMMA GEMM, both directions)
  gx_gemm_kernel<<<dim3(Tn * Bn / 64, G3H / 256, 2), 256, 0, stream>>>(
      xb, wihB, bih_f, bih_b, gx);

  // 4) single persistent launch runs the whole 256-step bidirectional scan
  gru_scan_kernel<<<dim3(NBLK, 1, 2), 256, 0, stream>>>(
      whhB, bhh_f, bhh_b, gx, lengths, hF, hB, out, bar);

  // 5) final hidden states -> tail of d_out ([2][B][H] after outputs)
  final_h_kernel<<<(2 * Bn * Hn + 255) / 256, 256, 0, stream>>>(
      hF, out + (size_t)Tn * Bn * 2 * Hn);
}
